// SheafLearner_65352222376000
// MI455X (gfx1250) — matrix operand order repaired and checked by reading, not compiled, not run on hardware
//
#include <hip/hip_runtime.h>
#include <hip/hip_fp16.h>

typedef __attribute__((ext_vector_type(16))) _Float16 v16h;
typedef __attribute__((ext_vector_type(8)))  float    v8f;
typedef __attribute__((ext_vector_type(4)))  _Float16 h4;

#define N_NODES 50000
#define N_EDGES 1600000
#define IN_FEAT 18
#define HID 64
#define EK_WAVES 4
#define EK_THREADS (EK_WAVES * 32)

// Branch-free tanh: tanh(x) = (e^{2x}-1) / (e^{2x}+1), clamped so exp never
// overflows. ~6 VALU ops (v_exp_f32 + v_rcp_f32), no EXEC divergence --
// replaces the ~30-instruction dual-branch libm tanhf seen in round-1 disasm.
__device__ __forceinline__ float fast_tanh(float x) {
  x = fminf(fmaxf(x, -15.f), 15.f);
  float e = __expf(2.f * x);
  return (e - 1.f) * __builtin_amdgcn_rcpf(e + 1.f);
}

// ---------------- encoder: h = relu(x @ We + be) ----------------
__global__ void enc_kernel(const float* __restrict__ x, const float* __restrict__ We,
                           const float* __restrict__ be, float* __restrict__ h) {
  int n = blockIdx.x * blockDim.x + threadIdx.x;
  if (n >= N_NODES) return;
  float xi[IN_FEAT];
#pragma unroll
  for (int i = 0; i < IN_FEAT; i++) xi[i] = x[n * IN_FEAT + i];
  for (int j = 0; j < HID; j++) {
    float s = be[j];
#pragma unroll
    for (int i = 0; i < IN_FEAT; i++) s += xi[i] * We[i * HID + j];
    h[n * HID + j] = s > 0.f ? s : 0.f;
  }
}

// ------- edge kernel: r = tanh(tanh([h_row||h_col]@W1+b1)@W2+b2); -------
// ------- scatter-add h_row * r into aggr[col] (WMMA f16 path)     -------
__global__ __launch_bounds__(EK_THREADS)
void edge_kernel(const float* __restrict__ h, const int* __restrict__ eidx,
                 const float* __restrict__ W1, const float* __restrict__ b1,
                 const float* __restrict__ W2, const float* __restrict__ b2p,
                 float* __restrict__ aggr) {
  __shared__ _Float16 w1s[128 * 32];     // W1 as f16, [K=128][N=32]
  __shared__ float    b1s[32], w2s[32];
  __shared__ float    b2s;
  __shared__ _Float16 ein[EK_WAVES][16 * 128];  // per-wave edge tile, f16
  __shared__ float    hid[EK_WAVES][16 * 32];   // per-wave hidden tile
  __shared__ float    rt[EK_WAVES][16];
  __shared__ int      rn[EK_WAVES][16], cn[EK_WAVES][16];

  const int tid  = threadIdx.x;
  const int wave = tid >> 5;
  const int lane = tid & 31;
  const int g    = lane >> 4;   // lane half (selects K-group in A/B layouts)
  const int l15  = lane & 15;

  // block-cooperative weight staging
  for (int i = tid; i < 128 * 32; i += EK_THREADS) w1s[i] = (_Float16)W1[i];
  if (tid < 32) { b1s[tid] = b1[tid]; w2s[tid] = W2[tid]; }
  if (tid == 0) b2s = b2p[0];
  __syncthreads();

  // Preload B fragments (32x16 f16 per (kc,ntile)), resident in VGPRs.
  // B layout (32x16 f16): lane n = L&15 (+16*ntile), K = g*16 + 2j + p.
  v16h bf[4][2];
#pragma unroll
  for (int kc = 0; kc < 4; kc++)
#pragma unroll
    for (int nt = 0; nt < 2; nt++) {
      int n = l15 + nt * 16;
#pragma unroll
      for (int e = 0; e < 16; e++) {
        int j = e >> 1, p = e & 1;
        int k = kc * 32 + g * 16 + 2 * j + p;
        bf[kc][nt][e] = w1s[k * 32 + n];
      }
    }

  // hoist per-lane constants out of the tile loop
  const float bias_nt[2] = { b1s[l15], b1s[l15 + 16] };
  const float b2v = b2s;

  const int ntile  = N_EDGES / 16;           // 100000 tiles, exact
  const int nwaves = gridDim.x * EK_WAVES;
  for (int t = blockIdx.x * EK_WAVES + wave; t < ntile; t += nwaves) {
    const int ebase = t * 16;

    // ---- stage 16 edges x 128 features into LDS as f16 ----
    {
      int e    = lane >> 1;
      int half = lane & 1;             // 0 -> row endpoint, 1 -> col endpoint
      int node = eidx[half * N_EDGES + ebase + e];
      if (half == 0) rn[wave][e] = node; else cn[wave][e] = node;
      const float4* hp = (const float4*)(h + node * HID);
      _Float16* dst = &ein[wave][e * 128 + half * 64];
#pragma unroll
      for (int q = 0; q < 16; q++) {
        float4 v = hp[q];
        h4 hv = {(_Float16)v.x, (_Float16)v.y, (_Float16)v.z, (_Float16)v.w};
        *(h4*)(dst + q * 4) = hv;      // ds_store_b64, 8B aligned
      }
    }
    asm volatile("s_wait_dscnt 0" ::: "memory");

    // ---- hidden = tanh(edge_in @ W1 + b1) via v_wmma_f32_16x16x32_f16 ----
#pragma unroll
    for (int nt = 0; nt < 2; nt++) {
      v8f c;
      float bias = bias_nt[nt];
#pragma unroll
      for (int j = 0; j < 8; j++) c[j] = bias;   // C init = bias (N = l15)
#pragma unroll
      for (int kc = 0; kc < 4; kc++) {
        // A layout (16x32 f16): lane m = L&15; K = g*16 + e
        v16h a;
#pragma unroll
        for (int e = 0; e < 16; e++) {
          int j = e >> 1, p = e & 1;
          int kl = g * 16 + e; (void)j; (void)p;
          a[e] = ein[wave][l15 * 128 + kc * 32 + kl];
        }
        c = __builtin_amdgcn_wmma_f32_16x16x32_f16(false, a, false, bf[kc][nt],
                                                   (short)0, c, false, false);
      }
      // D layout: M = j + 8*g, N = l15 + 16*nt
#pragma unroll
      for (int j = 0; j < 8; j++) {
        int m = j + 8 * g;
        hid[wave][m * 32 + l15 + nt * 16] = fast_tanh(c[j]);
      }
    }
    asm volatile("s_wait_dscnt 0" ::: "memory");

    // ---- r = tanh(hidden @ W2 + b2), one lane per edge ----
    if (lane < 16) {
      float s = b2v;
#pragma unroll
      for (int n = 0; n < 32; n++) s += hid[wave][lane * 32 + n] * w2s[n];
      rt[wave][lane] = fast_tanh(s);
    }
    asm volatile("s_wait_dscnt 0" ::: "memory");

    // ---- scatter: aggr[col] += h[row] * r (f32 atomics) ----
#pragma unroll 1
    for (int m = 0; m < 16; m++) {
      float r  = rt[wave][m];
      int   nr = rn[wave][m];
      int   nc = cn[wave][m];
      float v0 = h[nr * HID + lane]      * r;
      float v1 = h[nr * HID + lane + 32] * r;
      atomicAdd(&aggr[nc * HID + lane],      v0);
      atomicAdd(&aggr[nc * HID + lane + 32], v1);
    }
  }
}

// ---------------- residual diffusion update ----------------
__global__ void update_kernel(const float* __restrict__ hin, const float* __restrict__ aggr,
                              const float* __restrict__ alpha_p, float* __restrict__ hout,
                              int do_relu) {
  int i = blockIdx.x * blockDim.x + threadIdx.x;
  if (i >= N_NODES * HID) return;
  float a = alpha_p[0];
  float v = hin[i];
  float u = v - a * (v - aggr[i]);
  if (do_relu) u = u > 0.f ? u : 0.f;
  hout[i] = u;
}

// ---------------- decoder: relu(h@Wd1+bd1)@Wd2+bd2 ----------------
__global__ void dec_kernel(const float* __restrict__ h, const float* __restrict__ Wd1,
                           const float* __restrict__ bd1, const float* __restrict__ Wd2,
                           const float* __restrict__ bd2, float* __restrict__ out) {
  int n = blockIdx.x * blockDim.x + threadIdx.x;
  if (n >= N_NODES) return;
  float hi[HID];
#pragma unroll
  for (int i = 0; i < HID; i++) hi[i] = h[n * HID + i];
  float o0 = bd2[0], o1 = bd2[1];
  for (int j = 0; j < 32; j++) {
    float s = bd1[j];
#pragma unroll
    for (int i = 0; i < HID; i++) s += hi[i] * Wd1[i * 32 + j];
    s = s > 0.f ? s : 0.f;
    o0 += s * Wd2[j * 2 + 0];
    o1 += s * Wd2[j * 2 + 1];
  }
  out[n * 2 + 0] = o0;
  out[n * 2 + 1] = o1;
}

extern "C" void kernel_launch(void* const* d_in, const int* in_sizes, int n_in,
                              void* d_out, int out_size, void* d_ws, size_t ws_size,
                              hipStream_t stream) {
  (void)in_sizes; (void)n_in; (void)out_size; (void)ws_size;
  const float* x      = (const float*)d_in[0];
  const int*   eidx   = (const int*)  d_in[1];
  const float* We     = (const float*)d_in[2];
  const float* be     = (const float*)d_in[3];
  const float* s1_W1  = (const float*)d_in[4];
  const float* s1_b1  = (const float*)d_in[5];
  const float* s1_W2  = (const float*)d_in[6];
  const float* s1_b2  = (const float*)d_in[7];
  const float* s1_al  = (const float*)d_in[8];
  const float* s2_W1  = (const float*)d_in[9];
  const float* s2_b1  = (const float*)d_in[10];
  const float* s2_W2  = (const float*)d_in[11];
  const float* s2_b2  = (const float*)d_in[12];
  const float* s2_al  = (const float*)d_in[13];
  const float* Wd1    = (const float*)d_in[14];
  const float* bd1    = (const float*)d_in[15];
  const float* Wd2    = (const float*)d_in[16];
  const float* bd2    = (const float*)d_in[17];

  const size_t NH = (size_t)N_NODES * HID;
  float* hA   = (float*)d_ws;
  float* hB   = hA + NH;
  float* aggr = hB + NH;

  // h0 = relu(x @ We + be)
  enc_kernel<<<(N_NODES + 127) / 128, 128, 0, stream>>>(x, We, be, hA);

  // sheaf layer 1 (into hB, then relu)
  hipMemsetAsync(aggr, 0, NH * sizeof(float), stream);
  edge_kernel<<<2048, EK_THREADS, 0, stream>>>(hA, eidx, s1_W1, s1_b1, s1_W2, s1_b2, aggr);
  update_kernel<<<(int)((NH + 255) / 256), 256, 0, stream>>>(hA, aggr, s1_al, hB, 1);

  // sheaf layer 2 (into hA, no relu)
  hipMemsetAsync(aggr, 0, NH * sizeof(float), stream);
  edge_kernel<<<2048, EK_THREADS, 0, stream>>>(hB, eidx, s2_W1, s2_b1, s2_W2, s2_b2, aggr);
  update_kernel<<<(int)((NH + 255) / 256), 256, 0, stream>>>(hB, aggr, s2_al, hA, 0);

  // decode
  dec_kernel<<<(N_NODES + 127) / 128, 128, 0, stream>>>(hA, Wd1, bd1, Wd2, bd2, (float*)d_out);
}
